// EfficientMultimodalCrossAttention_59622736003254
// MI455X (gfx1250) — matrix operand order repaired and checked by reading, not compiled, not run on hardware
//
#include <hip/hip_runtime.h>
#include <hip/hip_bf16.h>

// ---------------------------------------------------------------------------
// EfficientMultimodalCrossAttention for MI455X (gfx1250).
// bf16 WMMA GEMMs with async-to-LDS double-buffered staging, wave-per-query
// top-k attention using the re-softmax identity:
//   o = (sum_all V + sum_topk (exp(w_j)-1) V_j) / (sum_topk exp(w_j) + S-K)
// ---------------------------------------------------------------------------

typedef __attribute__((ext_vector_type(16))) __bf16 v16bf;
typedef __attribute__((ext_vector_type(8)))  float  v8f;

#define D_MODEL 1024
#define NH      16
#define HD      64
#define TOTS    1536          // concatenated KV sequence (1024 text + 256 + 256)
#define BATCH   2
#define NROWS   (BATCH * TOTS)   // 3072 activation rows
#define TOPK_N  32
#define QK_SCALE 0.125f       // 64^-0.5
#define NEG_INF (-3.0e38f)
#define STR     1040          // LDS score row stride (floats), conflict-avoiding

#define USE_ASYNC 1           // gfx1250 GLOBAL_LOAD_ASYNC_TO_LDS path

// ---- async 16B global->LDS copy (one lane = one b128 transfer) -------------
__device__ inline void async_cp16(const __bf16* g, __bf16* lds_generic) {
#if USE_ASYNC
  unsigned int lds_off = (unsigned int)(size_t)lds_generic;  // flat->LDS aperture: addr[31:0]
  unsigned long long ga = (unsigned long long)(size_t)g;
  asm volatile("global_load_async_to_lds_b128 %0, %1, off"
               :: "v"(lds_off), "v"(ga) : "memory");
#else
  *(uint4*)lds_generic = *(const uint4*)g;
#endif
}
__device__ inline void async_wait_all() {
#if USE_ASYNC
  asm volatile("s_wait_asynccnt 0x0" ::: "memory");
#endif
}

// ---- WMMA fragment loaders (ISA 7.12.2 layouts, wave32) --------------------
// A (16x32 bf16, MxK): lane m = lane&15 holds row m; lanes 0-15 cover K lo
// halves (0..7,16..23), lanes 16-31 the hi halves (8..15,24..31).
__device__ inline v16bf load_a_frag(const __bf16* rowbase, int lane, int koff) {
  int hi = lane >> 4;
  v16bf a;
#pragma unroll
  for (int j = 0; j < 8; ++j) {
    int kk = 2 * (j & 3) + 8 * hi + 16 * (j >> 2) + koff;
    a[2 * j]     = rowbase[kk];
    a[2 * j + 1] = rowbase[kk + 1];
  }
  return a;
}
// B (32x16 bf16, KxN) built from a row-major [N][K] matrix: lane n = lane&15,
// same K order as A: lanes 0-15 K (0..7,16..23), lanes 16-31 K (8..15,24..31).
__device__ inline v16bf load_b_frag(const __bf16* colbase, int lane, int koff) {
  int hi = lane >> 4;
  v16bf b;
#pragma unroll
  for (int j = 0; j < 8; ++j) {
    int kk = 2 * (j & 3) + 8 * hi + 16 * (j >> 2) + koff;
    b[2 * j]     = colbase[kk];
    b[2 * j + 1] = colbase[kk + 1];
  }
  return b;
}

// ---- 1) LayerNorm (fp32 in) -> bf16 concatenated activations ---------------
__global__ void ln_to_bf16_kernel(const float* __restrict__ xt,
                                  const float* __restrict__ xi,
                                  const float* __restrict__ xa,
                                  const float* __restrict__ g,
                                  const float* __restrict__ be,
                                  __bf16* __restrict__ xn) {
  int row = blockIdx.x;               // 0..NROWS-1
  int b = row / TOTS, s = row % TOTS;
  int mod;
  const float* src;
  if (s < 1024)      { mod = 0; src = xt + ((size_t)b * 1024 + s) * D_MODEL; }
  else if (s < 1280) { mod = 1; src = xi + ((size_t)b * 256 + (s - 1024)) * D_MODEL; }
  else               { mod = 2; src = xa + ((size_t)b * 256 + (s - 1280)) * D_MODEL; }

  __shared__ float red[256];
  int tid = threadIdx.x;
  float v[4], lsum = 0.f, lsq = 0.f;
#pragma unroll
  for (int i = 0; i < 4; ++i) {
    v[i] = src[tid + 256 * i];
    lsum += v[i];
    lsq  += v[i] * v[i];
  }
  red[tid] = lsum; __syncthreads();
  for (int o = 128; o; o >>= 1) { if (tid < o) red[tid] += red[tid + o]; __syncthreads(); }
  float mu = red[0] * (1.f / D_MODEL);
  __syncthreads();
  red[tid] = lsq; __syncthreads();
  for (int o = 128; o; o >>= 1) { if (tid < o) red[tid] += red[tid + o]; __syncthreads(); }
  float var = red[0] * (1.f / D_MODEL) - mu * mu;
  float rs = rsqrtf(var + 1e-5f);

  __bf16* dst = xn + (size_t)row * D_MODEL;
#pragma unroll
  for (int i = 0; i < 4; ++i) {
    int d = tid + 256 * i;
    float y = (v[i] - mu) * rs * g[mod * D_MODEL + d] + be[mod * D_MODEL + d];
    dst[d] = (__bf16)y;
  }
}

// ---- 2) fp32 -> bf16 weight conversion -------------------------------------
__global__ void f2bf_kernel(const float* __restrict__ s, __bf16* __restrict__ d, int n) {
  int i = blockIdx.x * 256 + threadIdx.x;
  if (i < n) d[i] = (__bf16)s[i];
}

// ---- 3) QKV projection: xn @ W[mod]^T + bias, head-major output ------------
// Block = 8 waves, block tile 64(M) x 256(N); wave tile 32x64 (2x4 subtiles).
// A tile (64x32 bf16 per k-step) staged to LDS with async DMA, double-buffered.
__global__ void qkv_proj_kernel(const __bf16* __restrict__ xn,
                                const __bf16* __restrict__ Wq,
                                const __bf16* __restrict__ Wk,
                                const __bf16* __restrict__ Wv,
                                const float* __restrict__ bq,
                                const float* __restrict__ bk,
                                const float* __restrict__ bv,
                                __bf16* __restrict__ Qb,
                                __bf16* __restrict__ Kb,
                                __bf16* __restrict__ Vb) {
  int which = blockIdx.z;
  const __bf16* W  = (which == 0) ? Wq : (which == 1) ? Wk : Wv;
  const float*  bs = (which == 0) ? bq : (which == 1) ? bk : bv;
  __bf16*       dst = (which == 0) ? Qb : (which == 1) ? Kb : Vb;

  __shared__ __bf16 ldsA[2][64 * 32];          // double-buffered A tile

  int wave = threadIdx.x >> 5, lane = threadIdx.x & 31;
  int lo = lane & 15, hi = lane >> 4;
  int blk = blockIdx.x;                        // 0..191
  int mrow0 = (blk >> 2) * 64;                 // 48 M-blocks of 64 rows
  int n0 = (blk & 3) * 256;                    // 4 N-blocks of 256 cols
  int srow = mrow0 % TOTS;
  int mod = (srow < 1024) ? 0 : (srow < 1280) ? 1 : 2;

  int mw = wave & 1, nw = wave >> 1;           // wave -> (2 x 4) position
  const __bf16* Bb[4];
#pragma unroll
  for (int bt = 0; bt < 4; ++bt)
    Bb[bt] = W + ((size_t)mod * D_MODEL + n0 + nw * 64 + bt * 16 + lo) * D_MODEL;

  // staging assignment: thread -> (row 0..63, 16B chunk 0..3)
  int trow = threadIdx.x >> 2;
  int tchunk = (threadIdx.x & 3) * 8;          // element offset within 32-wide k slab
  const __bf16* Ag = xn + (size_t)(mrow0 + trow) * D_MODEL + tchunk;

  v8f c[2][4];
#pragma unroll
  for (int mt = 0; mt < 2; ++mt)
#pragma unroll
    for (int bt = 0; bt < 4; ++bt) c[mt][bt] = (v8f){};

  // prologue: stage k-step 0
  async_cp16(Ag, &ldsA[0][trow * 32 + tchunk]);
  async_wait_all();
  __syncthreads();

  for (int ks = 0; ks < D_MODEL; ks += 32) {
    int cur = (ks >> 5) & 1;
    if (ks + 32 < D_MODEL) {                   // stage next slab (overlaps compute)
      async_cp16(Ag + ks + 32, &ldsA[1 - cur][trow * 32 + tchunk]);
#pragma unroll
      for (int bt = 0; bt < 4; ++bt)           // prefetch next weight columns
        __builtin_prefetch(Bb[bt] + ks + 32, 0, 1);
    }
    v16bf a[2];
#pragma unroll
    for (int mt = 0; mt < 2; ++mt)
      a[mt] = load_a_frag(&ldsA[cur][(mw * 32 + mt * 16 + lo) * 32], lane, 0);
#pragma unroll
    for (int bt = 0; bt < 4; ++bt) {
      v16bf b = load_b_frag(Bb[bt], lane, ks);
#pragma unroll
      for (int mt = 0; mt < 2; ++mt)
        c[mt][bt] = __builtin_amdgcn_wmma_f32_16x16x32_bf16(false, a[mt], false, b,
                                                            (short)0, c[mt][bt], false, false);
    }
    async_wait_all();
    __syncthreads();
  }

  // epilogue: bias (+ Q scale), scatter to head-major [b,h,s,hd]
#pragma unroll
  for (int bt = 0; bt < 4; ++bt) {
    int n = n0 + nw * 64 + bt * 16 + lo;
    float bias = bs[mod * D_MODEL + n];
    int h = n >> 6, hd = n & 63;
#pragma unroll
    for (int mt = 0; mt < 2; ++mt) {
#pragma unroll
      for (int r = 0; r < 8; ++r) {
        int m = mrow0 + mw * 32 + mt * 16 + r + 8 * hi;
        int b_ = m / TOTS, s = m % TOTS;
        float val = c[mt][bt][r] + bias;
        if (which == 0) val *= QK_SCALE;
        dst[(((size_t)b_ * NH + h) * TOTS + s) * HD + hd] = (__bf16)val;
      }
    }
  }
}

// ---- 4) Vsum[b,h,hd] = sum over all 1536 keys of V -------------------------
__global__ void vsum_kernel(const __bf16* __restrict__ Vb, float* __restrict__ Vs) {
  int i = blockIdx.x * 256 + threadIdx.x;       // 0..2047
  if (i >= BATCH * NH * HD) return;
  int hd = i & 63, bh = i >> 6;
  const __bf16* p = Vb + (size_t)bh * TOTS * HD + hd;
  float s = 0.f;
  for (int k = 0; k < TOTS; ++k) s += (float)p[(size_t)k * HD];
  Vs[i] = s;
}

// ---- 5) Attention: WMMA scores -> masked softmax -> top-32 -> Vsum trick ---
// Block = 512 threads = 16 waves; handles one (b,h,modality,16-query tile).
__global__ void attn_kernel(const __bf16* __restrict__ Qb,
                            const __bf16* __restrict__ Kb,
                            const __bf16* __restrict__ Vb,
                            const float* __restrict__ Vs,
                            __bf16* __restrict__ obuf) {
  extern __shared__ float sc[];                 // [16][STR]
  int bid = blockIdx.x;
  int b = bid / (NH * 96);
  int rem = bid % (NH * 96);
  int h = rem / 96;
  int t = rem % 96;
  int qtile, mod_off;
  if (t < 64)      { qtile = t;      mod_off = 0;    }
  else if (t < 80) { qtile = t - 64; mod_off = 1024; }
  else             { qtile = t - 80; mod_off = 1280; }

  int wave = threadIdx.x >> 5, lane = threadIdx.x & 31;
  int hi = lane >> 4, lo = lane & 15;
  size_t bh = (size_t)b * NH + h;

  // ---- Phase A: score tiles via bf16 WMMA into LDS ----
  const __bf16* Qp = Qb + (bh * TOTS + mod_off + qtile * 16 + lo) * HD;
  v16bf a0 = load_a_frag(Qp, lane, 0);
  v16bf a1 = load_a_frag(Qp, lane, 32);
  int nkt = qtile + 1;                           // key tiles: 0..qtile (text region)
  for (int kt = wave; kt < nkt; kt += 16) {
    const __bf16* Kp = Kb + (bh * TOTS + kt * 16 + lo) * HD;
    v16bf b0 = load_b_frag(Kp, lane, 0);
    v16bf b1 = load_b_frag(Kp, lane, 32);
    v8f c = {};
    c = __builtin_amdgcn_wmma_f32_16x16x32_bf16(false, a0, false, b0, (short)0, c, false, false);
    c = __builtin_amdgcn_wmma_f32_16x16x32_bf16(false, a1, false, b1, (short)0, c, false, false);
#pragma unroll
    for (int r = 0; r < 8; ++r)
      sc[(r + 8 * hi) * STR + kt * 16 + lo] = c[r];
  }
  __syncthreads();

  // ---- Phase B: wave w owns query row r = w; lane owns cols k%32==lane ----
  int r = wave;
  int q = qtile * 16 + r;                        // local query index == mask bound
  int ncols = nkt * 16;
  float* row = sc + r * STR;

  float m = NEG_INF;
  for (int kk = lane; kk < ncols; kk += 32) {
    float s = row[kk];
    if (kk > q) { s = NEG_INF; row[kk] = s; }    // apply causal mask in place
    m = fmaxf(m, s);
  }
  for (int o = 16; o; o >>= 1) m = fmaxf(m, __shfl_xor(m, o, 32));

  float z1 = 0.f;
  for (int kk = lane; kk < ncols; kk += 32) z1 += __expf(row[kk] - m);
  for (int o = 16; o; o >>= 1) z1 += __shfl_xor(z1, o, 32);
  float inv_z1 = 1.f / z1;

  // top-32 selection + accumulation of (exp(w)-1)*V and Z
  float oacc0 = 0.f, oacc1 = 0.f, zacc = 0.f;
  int hd0 = lane * 2;
  for (int it = 0; it < TOPK_N; ++it) {
    float lv = NEG_INF;
    int li = 0x7fffffff;
    for (int kk = lane; kk < ncols; kk += 32) {
      float s = row[kk];
      if (s > lv) { lv = s; li = kk; }
    }
    float gv = lv; int gi = li;
    for (int o = 16; o; o >>= 1) {
      float ov = __shfl_xor(gv, o, 32);
      int   oi = __shfl_xor(gi, o, 32);
      if (ov > gv || (ov == gv && oi < gi)) { gv = ov; gi = oi; }
    }
    float w = (gv <= NEG_INF) ? 0.f : __expf(gv - m) * inv_z1;  // softmax prob
    float e = __expf(w);                                         // re-softmax numerator
    zacc += e;
    float coef = e - 1.f;
    if (gi != 0x7fffffff) {
      const __bf16* Vp = Vb + (bh * TOTS + gi) * HD + hd0;
      oacc0 += coef * (float)Vp[0];
      oacc1 += coef * (float)Vp[1];
      if ((gi & 31) == lane) row[gi] = NEG_INF;  // consume own column only
    }
  }

  float invZ = 1.f / (zacc + (float)(TOTS - TOPK_N));
  const float* vsp = Vs + bh * HD + hd0;
  float o0 = (vsp[0] + oacc0) * invZ;
  float o1 = (vsp[1] + oacc1) * invZ;
  __bf16* op = obuf + ((size_t)b * TOTS + mod_off + q) * D_MODEL + h * HD + hd0;
  op[0] = (__bf16)o0;
  op[1] = (__bf16)o1;
}

// ---- 6) Output projection: o @ Wo^T + bo -> fp32 d_out (tuple order) -------
// Same blocked structure as qkv_proj (64x256 block tile, async A staging).
__global__ void out_proj_kernel(const __bf16* __restrict__ obuf,
                                const __bf16* __restrict__ Wo,
                                const float* __restrict__ bo,
                                float* __restrict__ out) {
  __shared__ __bf16 ldsA[2][64 * 32];

  int wave = threadIdx.x >> 5, lane = threadIdx.x & 31;
  int lo = lane & 15, hi = lane >> 4;
  int blk = blockIdx.x;
  int mrow0 = (blk >> 2) * 64;
  int n0 = (blk & 3) * 256;
  int mw = wave & 1, nw = wave >> 1;

  const __bf16* Bb[4];
#pragma unroll
  for (int bt = 0; bt < 4; ++bt)
    Bb[bt] = Wo + (size_t)(n0 + nw * 64 + bt * 16 + lo) * D_MODEL;

  int trow = threadIdx.x >> 2;
  int tchunk = (threadIdx.x & 3) * 8;
  const __bf16* Ag = obuf + (size_t)(mrow0 + trow) * D_MODEL + tchunk;

  v8f c[2][4];
#pragma unroll
  for (int mt = 0; mt < 2; ++mt)
#pragma unroll
    for (int bt = 0; bt < 4; ++bt) c[mt][bt] = (v8f){};

  async_cp16(Ag, &ldsA[0][trow * 32 + tchunk]);
  async_wait_all();
  __syncthreads();

  for (int ks = 0; ks < D_MODEL; ks += 32) {
    int cur = (ks >> 5) & 1;
    if (ks + 32 < D_MODEL) {
      async_cp16(Ag + ks + 32, &ldsA[1 - cur][trow * 32 + tchunk]);
#pragma unroll
      for (int bt = 0; bt < 4; ++bt)
        __builtin_prefetch(Bb[bt] + ks + 32, 0, 1);
    }
    v16bf a[2];
#pragma unroll
    for (int mt = 0; mt < 2; ++mt)
      a[mt] = load_a_frag(&ldsA[cur][(mw * 32 + mt * 16 + lo) * 32], lane, 0);
#pragma unroll
    for (int bt = 0; bt < 4; ++bt) {
      v16bf b = load_b_frag(Bb[bt], lane, ks);
#pragma unroll
      for (int mt = 0; mt < 2; ++mt)
        c[mt][bt] = __builtin_amdgcn_wmma_f32_16x16x32_bf16(false, a[mt], false, b,
                                                            (short)0, c[mt][bt], false, false);
    }
    async_wait_all();
    __syncthreads();
  }

  const size_t TXT = (size_t)BATCH * 1024 * D_MODEL;
  const size_t IMG = (size_t)BATCH * 256 * D_MODEL;
#pragma unroll
  for (int bt = 0; bt < 4; ++bt) {
    int n = n0 + nw * 64 + bt * 16 + lo;
    float bias = bo[n];
#pragma unroll
    for (int mt = 0; mt < 2; ++mt) {
#pragma unroll
      for (int r = 0; r < 8; ++r) {
        int m = mrow0 + mw * 32 + mt * 16 + r + 8 * hi;
        int b_ = m / TOTS, s = m % TOTS;
        size_t idx;
        if (s < 1024)      idx = ((size_t)b_ * 1024 + s) * D_MODEL + n;
        else if (s < 1280) idx = TXT + ((size_t)b_ * 256 + (s - 1024)) * D_MODEL + n;
        else               idx = TXT + IMG + ((size_t)b_ * 256 + (s - 1280)) * D_MODEL + n;
        out[idx] = c[mt][bt][r] + bias;
      }
    }
  }
}

// ---------------------------------------------------------------------------
extern "C" void kernel_launch(void* const* d_in, const int* in_sizes, int n_in,
                              void* d_out, int out_size, void* d_ws, size_t ws_size,
                              hipStream_t stream) {
  (void)in_sizes; (void)n_in; (void)out_size; (void)ws_size;
  const float* x_text  = (const float*)d_in[0];
  const float* x_image = (const float*)d_in[1];
  const float* x_audio = (const float*)d_in[2];
  const float* ln_g    = (const float*)d_in[3];
  const float* ln_b    = (const float*)d_in[4];
  const float* Wq      = (const float*)d_in[5];
  const float* bq      = (const float*)d_in[6];
  const float* Wk      = (const float*)d_in[7];
  const float* bk      = (const float*)d_in[8];
  const float* Wv      = (const float*)d_in[9];
  const float* bv      = (const float*)d_in[10];
  const float* Wo      = (const float*)d_in[11];
  const float* bo      = (const float*)d_in[12];
  float* out = (float*)d_out;

  // workspace partition (256B aligned)
  char* ws = (char*)d_ws;
  auto alloc = [&](size_t bytes) -> char* {
    char* p = ws;
    ws += (bytes + 255) & ~(size_t)255;
    return p;
  };
  const size_t DD = (size_t)D_MODEL * D_MODEL;
  __bf16* xn  = (__bf16*)alloc((size_t)NROWS * D_MODEL * 2);
  __bf16* Wqb = (__bf16*)alloc(3 * DD * 2);
  __bf16* Wkb = (__bf16*)alloc(3 * DD * 2);
  __bf16* Wvb = (__bf16*)alloc(3 * DD * 2);
  __bf16* Wob = (__bf16*)alloc(DD * 2);
  __bf16* Qb  = (__bf16*)alloc((size_t)BATCH * NH * TOTS * HD * 2);
  __bf16* Kb  = (__bf16*)alloc((size_t)BATCH * NH * TOTS * HD * 2);
  __bf16* Vb  = (__bf16*)alloc((size_t)BATCH * NH * TOTS * HD * 2);
  float*  Vsm = (float*)alloc((size_t)BATCH * NH * HD * 4);
  __bf16* ob  = (__bf16*)alloc((size_t)NROWS * D_MODEL * 2);

  // 1) LayerNorm -> bf16
  ln_to_bf16_kernel<<<NROWS, 256, 0, stream>>>(x_text, x_image, x_audio, ln_g, ln_b, xn);

  // 2) weight conversion
  int n3 = (int)(3 * DD);
  f2bf_kernel<<<(n3 + 255) / 256, 256, 0, stream>>>(Wq, Wqb, n3);
  f2bf_kernel<<<(n3 + 255) / 256, 256, 0, stream>>>(Wk, Wkb, n3);
  f2bf_kernel<<<(n3 + 255) / 256, 256, 0, stream>>>(Wv, Wvb, n3);
  f2bf_kernel<<<((int)DD + 255) / 256, 256, 0, stream>>>(Wo, Wob, (int)DD);

  // 3) QKV projection: 192 blocks x 3 matrices, 8 waves each (64x256 tiles)
  qkv_proj_kernel<<<dim3(192, 1, 3), 256, 0, stream>>>(xn, Wqb, Wkb, Wvb, bq, bk, bv, Qb, Kb, Vb);

  // 4) V column sums
  vsum_kernel<<<(BATCH * NH * HD + 255) / 256, 256, 0, stream>>>(Vb, Vsm);

  // 5) attention (96 query tiles per (b,h): 64 text + 16 image + 16 audio)
  attn_kernel<<<BATCH * NH * 96, 512, 16 * STR * sizeof(float), stream>>>(Qb, Kb, Vb, Vsm, ob);

  // 6) output projection straight into d_out
  out_proj_kernel<<<192, 256, 0, stream>>>(ob, Wob, bo, out);
}
